// MMOE_87402584473999
// MI455X (gfx1250) — matrix operand order repaired and checked by reading, not compiled, not run on hardware
//
#include <hip/hip_runtime.h>
#include <hip/hip_bf16.h>

typedef __attribute__((ext_vector_type(16))) __bf16 v16bf;
typedef __attribute__((ext_vector_type(8)))  float  v8f;

#define B_   16384
#define V_   200000
#define E_   8
#define H_   64
#define O_   32
#define TH_  32
#define T_   2

__device__ __forceinline__ unsigned short f32_to_bf16(float f) {
    union { float f; unsigned u; } x; x.f = f;
    unsigned r = x.u + 0x7FFFu + ((x.u >> 16) & 1u);   // round-to-nearest-even
    return (unsigned short)(r >> 16);
}

union Frag16 { unsigned short s[16]; uint4 u[2]; v16bf v; };

// ---------------------------------------------------------------------------
// Kernel 1: CSR row offsets from sorted batch index
// ---------------------------------------------------------------------------
__global__ void build_row_ptr(const int* __restrict__ index, int nnz,
                              int* __restrict__ row_ptr) {
    int i = blockIdx.x * blockDim.x + threadIdx.x;
    if (i >= nnz) return;
    int cur  = index[i];
    int prev = (i == 0) ? -1 : index[i - 1];
    for (int b = prev + 1; b <= cur; ++b) row_ptr[b] = i;
    if (i == nnz - 1)
        for (int b = cur + 1; b <= B_; ++b) row_ptr[b] = nnz;
}

// ---------------------------------------------------------------------------
// Kernel 2: transpose + bf16-convert small weights so WMMA B-fragments are
// contiguous per lane.  W1T: [E][O][H], W0T: [T][TH][O]
// ---------------------------------------------------------------------------
__global__ void prep_weights(const float* __restrict__ w_e1,  // [E][H][O]
                             const float* __restrict__ w_t0,  // [T][O][TH]
                             unsigned short* __restrict__ W1T,
                             unsigned short* __restrict__ W0T) {
    int i = blockIdx.x * blockDim.x + threadIdx.x;
    if (i < E_ * O_ * H_) {
        int e = i / (O_ * H_);
        int r = i % (O_ * H_);
        int n = r / H_, k = r % H_;
        W1T[i] = f32_to_bf16(w_e1[((size_t)e * H_ + k) * O_ + n]);
    }
    if (i < T_ * TH_ * O_) {
        int t = i / (TH_ * O_);
        int r = i % (TH_ * O_);
        int n = r / O_, k = r % O_;
        W0T[i] = f32_to_bf16(w_t0[((size_t)t * O_ + k) * TH_ + n]);
    }
}

// ---------------------------------------------------------------------------
// Kernel 3 (bandwidth-dominant): gather + segment-sum for experts & gates.
// One block (128 threads) per batch row; thread = (expert e, float4 group q);
// each nnz costs one b128 load + 4 FMAs per thread.  (feat,val) chunks staged
// through LDS.  Writes relu(h) as packed bf16x4 + softmaxed gates as f32.
// ---------------------------------------------------------------------------
__global__ __launch_bounds__(128) void stageA(
        const int*   __restrict__ row_ptr,
        const int*   __restrict__ feats,
        const float* __restrict__ values,
        const float* __restrict__ w_e0,     // [E][V][H]
        const float* __restrict__ w_g,      // [T][V][E]
        unsigned short* __restrict__ hrelu, // [B][E*H] bf16
        float* __restrict__ gates)          // [B][T*E]
{
    __shared__ int   s_feat[128];
    __shared__ float s_val[128];
    __shared__ float s_gl[T_ * E_];

    const int b   = blockIdx.x;
    const int tid = threadIdx.x;
    const int e   = tid >> 4;        // 0..7
    const int q   = tid & 15;        // float4 group within H=64
    const int gt  = tid >> 3;        // task   for tid<16
    const int ge  = tid & 7;         // expert for tid<16

    const int start = row_ptr[b];
    const int end   = row_ptr[b + 1];

    const float* wbase = w_e0 + (size_t)e * V_ * H_ + q * 4;

    float4 acc = make_float4(0.f, 0.f, 0.f, 0.f);
    float  gacc = 0.f;
    for (int k0 = start; k0 < end; k0 += 128) {
        const int cnt = (end - k0 < 128) ? (end - k0) : 128;
        __syncthreads();
        if (tid < cnt) { s_feat[tid] = feats[k0 + tid]; s_val[tid] = values[k0 + tid]; }
        __syncthreads();
        #pragma unroll 4
        for (int j = 0; j < cnt; ++j) {
            const int   f = s_feat[j];
            const float v = s_val[j];
            const float4 w = *(const float4*)(wbase + (size_t)f * H_);
            acc.x = fmaf(w.x, v, acc.x);
            acc.y = fmaf(w.y, v, acc.y);
            acc.z = fmaf(w.z, v, acc.z);
            acc.w = fmaf(w.w, v, acc.w);
            if (tid < 16)
                gacc = fmaf(w_g[((size_t)gt * V_ + f) * E_ + ge], v, gacc);
        }
    }
    // pack 4 bf16 into one 8-byte store
    union { unsigned short s[4]; uint2 u; } pk;
    pk.s[0] = f32_to_bf16(fmaxf(acc.x, 0.f));
    pk.s[1] = f32_to_bf16(fmaxf(acc.y, 0.f));
    pk.s[2] = f32_to_bf16(fmaxf(acc.z, 0.f));
    pk.s[3] = f32_to_bf16(fmaxf(acc.w, 0.f));
    *(uint2*)(hrelu + (size_t)b * (E_ * H_) + tid * 4) = pk.u;

    if (tid < 16) s_gl[tid] = gacc;
    __syncthreads();
    if (tid < T_) {
        float m = -3.4e38f;
        for (int i = 0; i < E_; ++i) m = fmaxf(m, s_gl[tid * E_ + i]);
        float ex[E_], sum = 0.f;
        for (int i = 0; i < E_; ++i) { ex[i] = __expf(s_gl[tid * E_ + i] - m); sum += ex[i]; }
        const float inv = 1.f / sum;
        for (int i = 0; i < E_; ++i)
            gates[(size_t)b * (T_ * E_) + tid * E_ + i] = ex[i] * inv;
    }
}

// ---------------------------------------------------------------------------
// Kernel 4: fully fused dense path on WMMA.  One wave32 per 16-row tile:
// 8 expert GEMMs (v_wmma_f32_16x16x32_bf16, 2 k-steps x 2 n-tiles each),
// gate-weighted mix on C-fragments, tower layer-1 via WMMA, final dot via
// shfl_xor butterfly.
// ---------------------------------------------------------------------------
__global__ __launch_bounds__(256) void stageB(
        const unsigned short* __restrict__ hrelu,  // [B][E*H] bf16
        const unsigned short* __restrict__ W1T,    // [E][O][H] bf16
        const float*          __restrict__ gates,  // [B][T*E]
        const unsigned short* __restrict__ W0T,    // [T][TH][O] bf16
        const float*          __restrict__ w_t1,   // [T][TH]
        float*                __restrict__ out)    // [B][T]
{
    __shared__ float s_g[8][16 * T_ * E_];   // gates tile per wave
    __shared__ float s_tin[8][16 * 32];      // towers_input relayout per wave

    const int lane = threadIdx.x & 31;
    const int wv   = threadIdx.x >> 5;
    const int tile = blockIdx.x * 8 + wv;
    const int row0 = tile * 16;
    const int rsel = lane & 15;              // row (A) or column (B/C)
    const int hsel = lane >> 4;              // lane-half

    // wave-local gate staging
    for (int i = lane; i < 16 * T_ * E_; i += 32)
        s_g[wv][i] = gates[(size_t)row0 * (T_ * E_) + i];
    asm volatile("s_wait_dscnt 0x0" ::: "memory");

    v8f tin[T_][2];
    for (int t = 0; t < T_; ++t)
        for (int n = 0; n < 2; ++n)
            for (int r = 0; r < 8; ++r) tin[t][n][r] = 0.f;

    const unsigned short* hb = hrelu + (size_t)row0 * (E_ * H_);

    for (int e = 0; e < E_; ++e) {
        // A fragments for K = e*64 .. e*64+63 (two 32-wide k-steps)
        Frag16 a0, a1;
        {
            const unsigned short* p = hb + (size_t)rsel * (E_ * H_) + e * H_ + hsel * 8;
            a0.u[0] = *(const uint4*)(p);
            a0.u[1] = *(const uint4*)(p + 16);
            a1.u[0] = *(const uint4*)(p + 32);
            a1.u[1] = *(const uint4*)(p + 48);
        }
        // per-row gate scales for this expert (C rows: r + 8*hsel)
        float gv[T_][8];
        for (int t = 0; t < T_; ++t)
            for (int r = 0; r < 8; ++r)
                gv[t][r] = s_g[wv][(r + 8 * hsel) * (T_ * E_) + t * E_ + e];

        for (int n = 0; n < 2; ++n) {
            Frag16 b0, b1;
            const unsigned short* q =
                W1T + ((size_t)e * O_ + n * 16 + rsel) * H_ + hsel * 8;
            b0.u[0] = *(const uint4*)(q);
            b0.u[1] = *(const uint4*)(q + 16);
            b1.u[0] = *(const uint4*)(q + 32);
            b1.u[1] = *(const uint4*)(q + 48);

            v8f c = {0.f, 0.f, 0.f, 0.f, 0.f, 0.f, 0.f, 0.f};
            c = __builtin_amdgcn_wmma_f32_16x16x32_bf16(false, a0.v, false, b0.v,
                                                        (short)0, c, false, false);
            c = __builtin_amdgcn_wmma_f32_16x16x32_bf16(false, a1.v, false, b1.v,
                                                        (short)0, c, false, false);
            for (int t = 0; t < T_; ++t)
                for (int r = 0; r < 8; ++r)
                    tin[t][n][r] += gv[t][r] * c[r];
        }
    }

    // ---- towers ----
    float res[T_][8];
    for (int t = 0; t < T_; ++t) {
        // relayout towers_input (C layout) -> LDS row-major [16][32]
        for (int r = 0; r < 8; ++r) {
            const int m = r + 8 * hsel;
            s_tin[wv][m * 32 + rsel]      = tin[t][0][r];
            s_tin[wv][m * 32 + 16 + rsel] = tin[t][1][r];
        }
        asm volatile("s_wait_dscnt 0x0" ::: "memory");

        // A fragment (bf16 convert from LDS); A 16x32 layout
        Frag16 a;
        for (int j = 0; j < 8; ++j) {
            a.s[j]     = f32_to_bf16(s_tin[wv][rsel * 32 + hsel * 8 + j]);
            a.s[8 + j] = f32_to_bf16(s_tin[wv][rsel * 32 + 16 + hsel * 8 + j]);
        }

        v8f ch[2];
        for (int n = 0; n < 2; ++n) {
            Frag16 bb;
            const unsigned short* q =
                W0T + ((size_t)t * TH_ + n * 16 + rsel) * O_ + hsel * 8;
            bb.u[0] = *(const uint4*)(q);
            bb.u[1] = *(const uint4*)(q + 16);
            v8f c = {0.f, 0.f, 0.f, 0.f, 0.f, 0.f, 0.f, 0.f};
            c = __builtin_amdgcn_wmma_f32_16x16x32_bf16(false, a.v, false, bb.v,
                                                        (short)0, c, false, false);
            for (int r = 0; r < 8; ++r) c[r] = fmaxf(c[r], 0.f);   // relu
            ch[n] = c;
        }
        const float wa = w_t1[t * TH_ + rsel];
        const float wb = w_t1[t * TH_ + 16 + rsel];
        for (int r = 0; r < 8; ++r) {
            float s = ch[0][r] * wa + ch[1][r] * wb;
            s += __shfl_xor(s, 1, 32);
            s += __shfl_xor(s, 2, 32);
            s += __shfl_xor(s, 4, 32);
            s += __shfl_xor(s, 8, 32);   // reduce across 16 lanes per half
            res[t][r] = s;
        }
    }
    if (rsel == 0) {
        for (int t = 0; t < T_; ++t)
            for (int r = 0; r < 8; ++r)
                out[(size_t)(row0 + r + 8 * hsel) * T_ + t] = res[t][r];
    }
}

// ---------------------------------------------------------------------------
extern "C" void kernel_launch(void* const* d_in, const int* in_sizes, int n_in,
                              void* d_out, int out_size, void* d_ws, size_t ws_size,
                              hipStream_t stream) {
    const int*   index  = (const int*)d_in[1];
    const int*   feats  = (const int*)d_in[2];
    const float* values = (const float*)d_in[3];
    const float* w_e0   = (const float*)d_in[4];
    const float* w_e1   = (const float*)d_in[5];
    const float* w_g    = (const float*)d_in[6];
    const float* w_t0   = (const float*)d_in[7];
    const float* w_t1   = (const float*)d_in[8];
    float* out = (float*)d_out;
    const int nnz = in_sizes[1];

    char* ws = (char*)d_ws;
    size_t off = 0;
    auto take = [&](size_t bytes) -> void* {
        void* p = (void*)(ws + off);
        off += (bytes + 255) & ~(size_t)255;
        return p;
    };
    int*            row_ptr = (int*)take((size_t)(B_ + 1) * sizeof(int));
    unsigned short* hrelu   = (unsigned short*)take((size_t)B_ * E_ * H_ * 2);
    float*          gatesb  = (float*)take((size_t)B_ * T_ * E_ * sizeof(float));
    unsigned short* W1T     = (unsigned short*)take((size_t)E_ * O_ * H_ * 2);
    unsigned short* W0T     = (unsigned short*)take((size_t)T_ * TH_ * O_ * 2);
    (void)ws_size; (void)n_in; (void)out_size;

    build_row_ptr<<<(nnz + 255) / 256, 256, 0, stream>>>(index, nnz, row_ptr);
    prep_weights<<<(E_ * O_ * H_ + 255) / 256, 256, 0, stream>>>(w_e1, w_t0, W1T, W0T);
    stageA<<<B_, 128, 0, stream>>>(row_ptr, feats, values, w_e0, w_g, hrelu, gatesb);
    stageB<<<B_ / (16 * 8), 256, 0, stream>>>(hrelu, W1T, gatesb, W0T, w_t1, out);
}
